// NeuralODETrainer_50225347559896
// MI455X (gfx1250) — matrix operand order repaired and checked by reading, not compiled, not run on hardware
//
#include <hip/hip_runtime.h>
#include <hip/hip_bf16.h>

typedef __attribute__((ext_vector_type(16))) _Float16 v16h;
typedef __attribute__((ext_vector_type(8)))  _Float16 v8h;
typedef __attribute__((ext_vector_type(8)))  float    v8f;

#define DEVFN __device__ __forceinline__

namespace {

constexpr int BATCH   = 4096;
constexpr int NSTEPS  = 16;
constexpr int WAVES   = 2;               // waves (16-row tiles) per workgroup
constexpr int THREADS = WAVES * 32;
constexpr int ROWS_PER_BLOCK = WAVES * 16;
constexpr int NBLOCKS = BATCH / ROWS_PER_BLOCK;   // 128 workgroups

// ---- LDS layout (bytes) --------------------------------------------------
constexpr int SM_W1 = 0;                          // W1^T  [256][32]  f16 (K pad 13->32, zeros)
constexpr int SM_W2 = SM_W1 + 256*32*2;           // W2^T  [256][256] f16
constexpr int SM_W3 = SM_W2 + 256*256*2;          // W3^T  [128][256] f16
constexpr int SM_W4 = SM_W3 + 128*256*2;          // W4^T  [16][128]  f16 (N pad 12->16, zeros)
constexpr int SM_B1 = SM_W4 + 16*128*2;           // 256 f32
constexpr int SM_B2 = SM_B1 + 256*4;              // 256 f32
constexpr int SM_B3 = SM_B2 + 256*4;              // 128 f32
constexpr int SM_B4 = SM_B3 + 128*4;              // 16  f32
constexpr int SM_WAVE0 = SM_B4 + 16*4;
// per-wave region
constexpr int W_ACTA = 0;                         // [16][256] f16
constexpr int W_ACTB = W_ACTA + 16*256*2;         // [16][256] f16
constexpr int W_Y    = W_ACTB + 16*256*2;         // [16][16] f32 state
constexpr int W_YT   = W_Y + 16*16*4;             // [16][16] f32 stage input
constexpr int W_K    = W_YT + 16*16*4;            // 6 x [16][16] f32
constexpr int WAVE_BYTES = W_K + 6*16*16*4;       // 24576
constexpr int SMEM_TOTAL = SM_WAVE0 + WAVES*WAVE_BYTES;
static_assert(SMEM_TOTAL <= 320*1024, "LDS budget exceeded");

// ---- Dormand-Prince 5(4) tableau ----------------------------------------
constexpr float cA21 = 1.f/5.f;
constexpr float cA31 = 3.f/40.f,      cA32 = 9.f/40.f;
constexpr float cA41 = 44.f/45.f,     cA42 = -56.f/15.f,    cA43 = 32.f/9.f;
constexpr float cA51 = 19372.f/6561.f, cA52 = -25360.f/2187.f,
                cA53 = 64448.f/6561.f, cA54 = -212.f/729.f;
constexpr float cA61 = 9017.f/3168.f,  cA62 = -355.f/33.f,
                cA63 = 46732.f/5247.f, cA64 = 49.f/176.f,   cA65 = -5103.f/18656.f;
constexpr float cB1 = 35.f/384.f, cB3 = 500.f/1113.f, cB4 = 125.f/192.f,
                cB5 = -2187.f/6784.f, cB6 = 11.f/84.f;

// ---- WMMA helpers --------------------------------------------------------
DEVFN v8f wmma16(v16h a, v16h b, v8f c) {
  return __builtin_amdgcn_wmma_f32_16x16x32_f16(
      /*neg_a=*/false, a, /*neg_b=*/false, b,
      /*c_mod=*/(short)0, c, /*reuse_a=*/false, /*reuse_b=*/false);
}

DEVFN v16h pack16(v8h lo, v8h hi) {
  v16h r;
#pragma unroll
  for (int i = 0; i < 8; ++i) { r[i] = lo[i]; r[i + 8] = hi[i]; }
  return r;
}

// A-fragment (16x32 f16, row-striped): lanes 0-15 hold rows m, K=kb*32+[0..7]
// in V0-3 and K=kb*32+[16..23] in V4-7; lanes 16-31 shift K by +8.
DEVFN v16h load_a(const _Float16* act, int pitch, int lane, int kb) {
  const int m  = lane & 15;
  const int k0 = kb * 32 + ((lane >> 4) << 3);
  const v8h* p0 = reinterpret_cast<const v8h*>(act + m * pitch + k0);
  const v8h* p1 = reinterpret_cast<const v8h*>(act + m * pitch + k0 + 16);
  return pack16(*p0, *p1);
}

// B-fragment (32x16 f16): lane n = nbase + (lane&15); lanes 0-15 hold K=[0..7]
// and K=[16..23], lanes 16-31 shift K by +8, of the kb-th K-chunk. Weights stored [N][K]
// (transposed) so this is two runs of 8 contiguous f16 per lane.
DEVFN v16h load_b(const _Float16* wt, int pitch, int lane, int nbase, int kb) {
  const int n  = nbase + (lane & 15);
  const int k0 = kb * 32 + ((lane >> 4) << 3);
  const v8h* p = reinterpret_cast<const v8h*>(wt + n * pitch + k0);
  return pack16(p[0], p[2]);
}

// Fast SiLU: x * 1/(1+exp(-x)) with hardware v_exp_f32 + v_rcp_f32.
DEVFN float silu(float x) {
  return x * __builtin_amdgcn_rcpf(1.0f + __expf(-x));
}

struct Shared {
  const _Float16 *W1t, *W2t, *W3t, *W4t;
  const float *b1, *b2, *b3, *b4;
  _Float16 *actA, *actB;
};

// Hidden layer: D[16xNDIM] = silu( A[16xKDIM] * W^T + b ), f16 out to LDS.
// N-tile loop kept rolled (I$ residency); K-chunk loop fully unrolled
// (8 independent WMMAs + hoisted A fragments give the ILP).
template <int KDIM, int NDIM>
DEVFN void mlp_layer(const _Float16* actIn, int inPitch, const _Float16* Wt,
                     const float* bias, _Float16* actOut, int outPitch, int lane) {
  constexpr int NKB = KDIM / 32;
  v16h a[NKB];
#pragma unroll
  for (int kb = 0; kb < NKB; ++kb) a[kb] = load_a(actIn, inPitch, lane, kb);
  const int nl = lane & 15;
  const int m0 = (lane >> 4) << 3;
#pragma unroll 1
  for (int nt = 0; nt < NDIM / 16; ++nt) {
    const int n = nt * 16 + nl;
    const float bv = bias[n];
    v8f c;
#pragma unroll
    for (int r = 0; r < 8; ++r) c[r] = bv;          // bias pre-loaded into acc
#pragma unroll
    for (int kb = 0; kb < NKB; ++kb)
      c = wmma16(a[kb], load_b(Wt, KDIM, lane, nt * 16, kb), c);
#pragma unroll
    for (int r = 0; r < 8; ++r)
      actOut[(m0 + r) * outPitch + n] = (_Float16)silu(c[r]);
  }
}

// Input layer: state yt[16][16] f32 (13 live cols, zero pad) -> h1[16][256].
DEVFN void layer_in(const float* yt, const _Float16* W1t, const float* bias,
                    _Float16* actOut, int lane) {
  const int m  = lane & 15;
  const int k0 = (lane >> 4) << 3;
  v16h a;
#pragma unroll
  for (int i = 0; i < 8; ++i) {
    a[i]     = (_Float16)yt[m * 16 + k0 + i];   // K = 0..15 (cols 13..15 are 0)
    a[i + 8] = (_Float16)0.f;                   // K = 16..31 pad
  }
  const int nl = lane & 15;
  const int m0 = (lane >> 4) << 3;
#pragma unroll 1
  for (int nt = 0; nt < 16; ++nt) {
    const int n = nt * 16 + nl;
    const float bv = bias[n];
    v8f c;
#pragma unroll
    for (int r = 0; r < 8; ++r) c[r] = bv;
    c = wmma16(a, load_b(W1t, 32, lane, nt * 16, 0), c);  // W1t K-pad is zeroed
#pragma unroll
    for (int r = 0; r < 8; ++r)
      actOut[(m0 + r) * 256 + n] = (_Float16)silu(c[r]);
  }
}

// Output layer: h3[16][128] -> k[16][16] f32 (cols >= 12 are exactly 0).
DEVFN void layer_out(const _Float16* actIn, const _Float16* W4t, const float* bias,
                     float* kdst, int lane) {
  v16h a[4];
#pragma unroll
  for (int kb = 0; kb < 4; ++kb) a[kb] = load_a(actIn, 256, lane, kb);
  const int n  = lane & 15;
  const int m0 = (lane >> 4) << 3;
  const float bv = bias[n];
  v8f c;
#pragma unroll
  for (int r = 0; r < 8; ++r) c[r] = bv;
#pragma unroll
  for (int kb = 0; kb < 4; ++kb)
    c = wmma16(a[kb], load_b(W4t, 128, lane, 0, kb), c);
#pragma unroll
  for (int r = 0; r < 8; ++r) kdst[(m0 + r) * 16 + n] = c[r];
}

DEVFN void feval(const Shared& S, const float* yt, float* kdst, int lane) {
  layer_in(yt, S.W1t, S.b1, S.actA, lane);                          __syncthreads();
  mlp_layer<256, 256>(S.actA, 256, S.W2t, S.b2, S.actB, 256, lane); __syncthreads();
  mlp_layer<256, 128>(S.actB, 256, S.W3t, S.b3, S.actA, 256, lane); __syncthreads();
  layer_out(S.actA, S.W4t, S.b4, kdst, lane);                       __syncthreads();
}

}  // namespace

__global__ __launch_bounds__(THREADS)
void node_dopri5_kernel(const float* __restrict__ y0, const float* __restrict__ tspan,
                        const float* __restrict__ W1, const float* __restrict__ b1,
                        const float* __restrict__ W2, const float* __restrict__ b2,
                        const float* __restrict__ W3, const float* __restrict__ b3,
                        const float* __restrict__ W4, const float* __restrict__ b4,
                        float* __restrict__ out) {
  extern __shared__ char smem[];
  _Float16* W1t = (_Float16*)(smem + SM_W1);
  _Float16* W2t = (_Float16*)(smem + SM_W2);
  _Float16* W3t = (_Float16*)(smem + SM_W3);
  _Float16* W4t = (_Float16*)(smem + SM_W4);
  float* b1s = (float*)(smem + SM_B1);
  float* b2s = (float*)(smem + SM_B2);
  float* b3s = (float*)(smem + SM_B3);
  float* b4s = (float*)(smem + SM_B4);

  const int tid = threadIdx.x;

  // ---- Stage weights once: transpose to [N][K], convert f32 -> f16 -------
  for (int i = tid; i < 256 * 32; i += THREADS) {
    int n = i >> 5, k = i & 31;
    W1t[i] = (_Float16)((k < 13) ? W1[k * 256 + n] : 0.f);
  }
  for (int i = tid; i < 256 * 256; i += THREADS) {
    int n = i >> 8, k = i & 255;
    W2t[i] = (_Float16)W2[k * 256 + n];
  }
  for (int i = tid; i < 128 * 256; i += THREADS) {
    int n = i >> 8, k = i & 255;
    W3t[i] = (_Float16)W3[k * 128 + n];
  }
  for (int i = tid; i < 16 * 128; i += THREADS) {
    int n = i >> 7, k = i & 127;
    W4t[i] = (_Float16)((n < 12) ? W4[k * 12 + n] : 0.f);
  }
  for (int i = tid; i < 256; i += THREADS) b1s[i] = b1[i];
  for (int i = tid; i < 256; i += THREADS) b2s[i] = b2[i];
  for (int i = tid; i < 128; i += THREADS) b3s[i] = b3[i];
  for (int i = tid; i < 16;  i += THREADS) b4s[i] = (i < 12) ? b4[i] : 0.f;
  __syncthreads();

  const int wave = tid >> 5;
  const int lane = tid & 31;
  char* wbase = smem + SM_WAVE0 + wave * WAVE_BYTES;
  _Float16* actA = (_Float16*)(wbase + W_ACTA);
  _Float16* actB = (_Float16*)(wbase + W_ACTB);
  float* yv = (float*)(wbase + W_Y);
  float* yt = (float*)(wbase + W_YT);
  float* k1 = (float*)(wbase + W_K);
  float* k2 = k1 + 256;
  float* k3 = k1 + 512;
  float* k4 = k1 + 768;
  float* k5 = k1 + 1024;
  float* k6 = k1 + 1280;

  const Shared S{W1t, W2t, W3t, W4t, b1s, b2s, b3s, b4s, actA, actB};
  const int row0 = blockIdx.x * ROWS_PER_BLOCK + wave * 16;

  // ---- Load initial state (13 cols live, pad to 16 with zeros) -----------
  for (int e = lane; e < 256; e += 32) {
    int m = e >> 4, cc = e & 15;
    yv[e] = (cc < 13) ? y0[(row0 + m) * 13 + cc] : 0.f;
  }
  __syncthreads();
  // pred_y[0] = y0
  for (int e = lane; e < 16 * 13; e += 32) {
    int m = e / 13, cc = e - 13 * m;
    out[(size_t)(row0 + m) * 13 + cc] = yv[m * 16 + cc];
  }

  // ---- Time integration ---------------------------------------------------
#pragma unroll 1
  for (int s = 1; s < NSTEPS; ++s) {
    const float h = (tspan[s] - tspan[s - 1]) * 0.5f;  // SUBSTEPS == 2
#pragma unroll 1
    for (int sub = 0; sub < 2; ++sub) {
      for (int e = lane; e < 256; e += 32) yt[e] = yv[e];
      __syncthreads();
      feval(S, yt, k1, lane);
      for (int e = lane; e < 256; e += 32)
        yt[e] = yv[e] + h * (cA21 * k1[e]);
      __syncthreads();
      feval(S, yt, k2, lane);
      for (int e = lane; e < 256; e += 32)
        yt[e] = yv[e] + h * (cA31 * k1[e] + cA32 * k2[e]);
      __syncthreads();
      feval(S, yt, k3, lane);
      for (int e = lane; e < 256; e += 32)
        yt[e] = yv[e] + h * (cA41 * k1[e] + cA42 * k2[e] + cA43 * k3[e]);
      __syncthreads();
      feval(S, yt, k4, lane);
      for (int e = lane; e < 256; e += 32)
        yt[e] = yv[e] + h * (cA51 * k1[e] + cA52 * k2[e] + cA53 * k3[e] + cA54 * k4[e]);
      __syncthreads();
      feval(S, yt, k5, lane);
      for (int e = lane; e < 256; e += 32)
        yt[e] = yv[e] + h * (cA61 * k1[e] + cA62 * k2[e] + cA63 * k3[e] +
                             cA64 * k4[e] + cA65 * k5[e]);
      __syncthreads();
      feval(S, yt, k6, lane);
      for (int e = lane; e < 256; e += 32)
        yv[e] += h * (cB1 * k1[e] + cB3 * k3[e] + cB4 * k4[e] +
                      cB5 * k5[e] + cB6 * k6[e]);
      __syncthreads();
    }
    for (int e = lane; e < 16 * 13; e += 32) {
      int m = e / 13, cc = e - 13 * m;
      out[((size_t)s * BATCH + row0 + m) * 13 + cc] = yv[m * 16 + cc];
    }
    __syncthreads();
  }
}

extern "C" void kernel_launch(void* const* d_in, const int* in_sizes, int n_in,
                              void* d_out, int out_size, void* d_ws, size_t ws_size,
                              hipStream_t stream) {
  (void)in_sizes; (void)n_in; (void)out_size; (void)d_ws; (void)ws_size;
  const float* y0 = (const float*)d_in[0];
  const float* t  = (const float*)d_in[1];
  const float* W1 = (const float*)d_in[2];
  const float* b1 = (const float*)d_in[3];
  const float* W2 = (const float*)d_in[4];
  const float* b2 = (const float*)d_in[5];
  const float* W3 = (const float*)d_in[6];
  const float* b3 = (const float*)d_in[7];
  const float* W4 = (const float*)d_in[8];
  const float* b4 = (const float*)d_in[9];
  float* out = (float*)d_out;

  node_dopri5_kernel<<<dim3(NBLOCKS), dim3(THREADS), SMEM_TOTAL, stream>>>(
      y0, t, W1, b1, W2, b2, W3, b3, W4, b4, out);
}
